// GraphSAGE_GAT_5738076307734
// MI455X (gfx1250) — matrix operand order repaired and checked by reading, not compiled, not run on hardware
//
#include <hip/hip_runtime.h>
#include <hip/hip_bf16.h>

// Problem constants (from reference)
#define NN      16384
#define EE      262144
#define IN_DIM  512
#define HID     256
#define HEADS   4
#define BB      64
#define ACTIONS 32
#define E2      (EE + NN)
#define NEG_SLOPE 0.2f

typedef __attribute__((ext_vector_type(16))) __bf16 v16bf;
typedef __attribute__((ext_vector_type(8)))  float  v8f;

__device__ __forceinline__ __bf16 f2bf(float f) { return (__bf16)f; }

__device__ __forceinline__ void atomicMaxFloat(float* addr, float val) {
  int bits = __float_as_int(val);
  if (bits >= 0) atomicMax((int*)addr, bits);
  else           atomicMin((unsigned int*)addr, (unsigned int)bits);
}

// ---------------------------------------------------------------- utilities
__global__ void fill_f32(float* __restrict__ p, int n, float v) {
  int i = blockIdx.x * blockDim.x + threadIdx.x;
  if (i < n) p[i] = v;
}

__global__ void cvt_bf16(const float* __restrict__ in, __bf16* __restrict__ out, int n) {
  int i = blockIdx.x * blockDim.x + threadIdx.x;
  if (i < n) out[i] = f2bf(in[i]);
}

// ------------------------------------------------- SAGE neighbor scatter-sum
// One block (128 threads) per edge: float4 per thread, atomic add into agg[dst].
__global__ __launch_bounds__(128) void sage_scatter(
    const float* __restrict__ x, const long long* __restrict__ ei,
    float* __restrict__ agg, float* __restrict__ deg) {
  int e = blockIdx.x;
  int s = (int)ei[e];
  int d = (int)ei[EE + e];
  float4 v = ((const float4*)(x + (size_t)s * IN_DIM))[threadIdx.x];
  float* a = agg + (size_t)d * IN_DIM + threadIdx.x * 4;
  atomicAdd(a + 0, v.x);
  atomicAdd(a + 1, v.y);
  atomicAdd(a + 2, v.z);
  atomicAdd(a + 3, v.w);
  if (threadIdx.x == 0) atomicAdd(&deg[d], 1.0f);
}

// ------------------------------------------------- WMMA B-tile LDS staging
// Two 32x16 bf16 B fragments (cols [col0,col0+16) and [col0+16,col0+32)),
// each swizzled so lane l reads its 16-element fragment contiguously:
//   element e of lane l is B[K][N], N = l&15, K = (e&8 ? 16 : 0) + (l>=16 ? 8 : 0) + (e&7)
// Writer thread handles one k-row, 4 consecutive n (one b64 global load);
// the 4 LDS targets are (n+i + ((k&8)<<1))*16 + (k&7) + ((k&16)>>1), i.e. stride 16 elements.
// 256 threads cover the full 32x32 B slab; straight-line (no loop).

// -------------------------------------------------------------- SAGE GEMM
// h = relu( bf16(agg/deg) @ W_l + b_l + bf16(x) @ W_r ), stored as bf16.
// Block = 256 threads = 8 waves; each wave owns a 16x32 strip (2 WMMA accums).
__global__ __launch_bounds__(256) void sage_gemm(
    const float* __restrict__ x, const float* __restrict__ agg,
    const float* __restrict__ deg, const float* __restrict__ bias,
    const __bf16* __restrict__ wl, const __bf16* __restrict__ wr,
    __bf16* __restrict__ h_out) {
  __shared__ __align__(32) __bf16 ldsB[1024];
  const int wave = threadIdx.x >> 5;
  const int lane = threadIdx.x & 31;
  const int row0 = blockIdx.x * 128 + wave * 16;
  const int col0 = blockIdx.y * 32;
  const int hi8  = (lane >> 4) << 3;          // 0 or 8 (A-fragment K phase)
  const int row  = row0 + (lane & 15);        // A row for this lane

  // B loader geometry (uniform, straight-line)
  const int bk  = threadIdx.x >> 3;           // k row   0..31
  const int bn  = (threadIdx.x & 7) * 4;      // n base  0..28 (4 cols)
  __bf16* bdst = &ldsB[(bn >> 4) * 512 + ((bn & 15) + ((bk & 8) << 1)) * 16 + (bk & 7) + ((bk & 16) >> 1)];

  float invdeg = 1.0f / fmaxf(deg[row], 1.0f);
  v8f acc0 = {}, acc1 = {};

  for (int pass = 0; pass < 2; ++pass) {
    const float*  Aptr  = pass ? x : agg;
    const __bf16* Bp    = pass ? wr : wl;
    const float   scale = pass ? 1.0f : invdeg;
    for (int k0 = 0; k0 < IN_DIM; k0 += 32) {
      __syncthreads();
      {
        ushort4 q = *(const ushort4*)(Bp + (size_t)(k0 + bk) * HID + col0 + bn);
        bdst[0]  = __builtin_bit_cast(__bf16, q.x);
        bdst[16] = __builtin_bit_cast(__bf16, q.y);
        bdst[32] = __builtin_bit_cast(__bf16, q.z);
        bdst[48] = __builtin_bit_cast(__bf16, q.w);
      }
      const float* ap = Aptr + (size_t)row * IN_DIM + k0 + hi8;
      __builtin_prefetch(ap + 32, 0, 3);      // WGP-scope prefetch of next K slice
      v16bf afrag;
      #pragma unroll
      for (int e = 0; e < 8; ++e) afrag[e]     = f2bf(ap[e]      * scale);
      #pragma unroll
      for (int e = 0; e < 8; ++e) afrag[8 + e] = f2bf(ap[16 + e] * scale);
      __syncthreads();
      v16bf b0 = *(const v16bf*)&ldsB[lane * 16];
      v16bf b1 = *(const v16bf*)&ldsB[512 + lane * 16];
      acc0 = __builtin_amdgcn_wmma_f32_16x16x32_bf16(false, afrag, false, b0,
                                                     (short)0, acc0, false, false);
      acc1 = __builtin_amdgcn_wmma_f32_16x16x32_bf16(false, afrag, false, b1,
                                                     (short)0, acc1, false, false);
    }
  }
  // C layout: VGPR i -> (M = i + hi8, N = lane&15)
  #pragma unroll
  for (int i = 0; i < 8; ++i) {
    int m = i + hi8, n = lane & 15;
    float v0 = fmaxf(acc0[i] + bias[col0 + n],      0.0f);
    float v1 = fmaxf(acc1[i] + bias[col0 + 16 + n], 0.0f);
    h_out[(size_t)(row0 + m) * HID + col0 + n]      = f2bf(v0);
    h_out[(size_t)(row0 + m) * HID + col0 + 16 + n] = f2bf(v1);
  }
}

// --------------------------------------------------------------- GAT GEMM
// g = h_bf16 @ gat_w_bf16   ([N,256] x [256,1024] -> [N,1024] fp32)
__global__ __launch_bounds__(256) void gat_gemm(
    const __bf16* __restrict__ hbf, const __bf16* __restrict__ wg,
    float* __restrict__ g) {
  __shared__ __align__(32) __bf16 ldsB[1024];
  const int wave = threadIdx.x >> 5;
  const int lane = threadIdx.x & 31;
  const int row0 = blockIdx.x * 128 + wave * 16;
  const int col0 = blockIdx.y * 32;
  const int hi8  = (lane >> 4) << 3;
  const int row  = row0 + (lane & 15);

  const int bk  = threadIdx.x >> 3;
  const int bn  = (threadIdx.x & 7) * 4;
  __bf16* bdst = &ldsB[(bn >> 4) * 512 + ((bn & 15) + ((bk & 8) << 1)) * 16 + (bk & 7) + ((bk & 16) >> 1)];

  v8f acc0 = {}, acc1 = {};
  for (int k0 = 0; k0 < HID; k0 += 32) {
    __syncthreads();
    {
      ushort4 q = *(const ushort4*)(wg + (size_t)(k0 + bk) * (HEADS * HID) + col0 + bn);
      bdst[0]  = __builtin_bit_cast(__bf16, q.x);
      bdst[16] = __builtin_bit_cast(__bf16, q.y);
      bdst[32] = __builtin_bit_cast(__bf16, q.z);
      bdst[48] = __builtin_bit_cast(__bf16, q.w);
    }
    const __bf16* ap = hbf + (size_t)row * HID + k0 + hi8;
    __builtin_prefetch(ap + 32, 0, 3);        // WGP-scope prefetch of next K slice
    v16bf afrag;
    #pragma unroll
    for (int e = 0; e < 8; ++e) afrag[e]     = ap[e];
    #pragma unroll
    for (int e = 0; e < 8; ++e) afrag[8 + e] = ap[16 + e];
    __syncthreads();
    v16bf b0 = *(const v16bf*)&ldsB[lane * 16];
    v16bf b1 = *(const v16bf*)&ldsB[512 + lane * 16];
    acc0 = __builtin_amdgcn_wmma_f32_16x16x32_bf16(false, afrag, false, b0,
                                                   (short)0, acc0, false, false);
    acc1 = __builtin_amdgcn_wmma_f32_16x16x32_bf16(false, afrag, false, b1,
                                                   (short)0, acc1, false, false);
  }
  #pragma unroll
  for (int i = 0; i < 8; ++i) {
    int m = i + hi8, n = lane & 15;
    g[(size_t)(row0 + m) * (HEADS * HID) + col0 + n]      = acc0[i];
    g[(size_t)(row0 + m) * (HEADS * HID) + col0 + 16 + n] = acc1[i];
  }
}

// ---------------------------------------- per-(node,head) attention logits
// One wave32 per (n, h): two 256-length dot products, shuffle reduction.
__global__ __launch_bounds__(256) void gat_att(
    const float* __restrict__ g, const float* __restrict__ att_src,
    const float* __restrict__ att_dst, float* __restrict__ a_src,
    float* __restrict__ a_dst) {
  int wid  = (blockIdx.x * blockDim.x + threadIdx.x) >> 5;
  int lane = threadIdx.x & 31;
  if (wid >= NN * HEADS) return;
  int n = wid >> 2, h = wid & 3;
  const float* gp = g + (size_t)n * (HEADS * HID) + h * HID;
  const float* as = att_src + h * HID;
  const float* ad = att_dst + h * HID;
  float s1 = 0.f, s2 = 0.f;
  #pragma unroll
  for (int c = lane; c < HID; c += 32) {
    float gv = gp[c];
    s1 += gv * as[c];
    s2 += gv * ad[c];
  }
  #pragma unroll
  for (int off = 16; off; off >>= 1) {
    s1 += __shfl_xor(s1, off, 32);
    s2 += __shfl_xor(s2, off, 32);
  }
  if (lane == 0) { a_src[wid] = s1; a_dst[wid] = s2; }
}

// ----------------------------------------------- edge softmax pass 1: max
__global__ __launch_bounds__(256) void edge_pass_max(
    const long long* __restrict__ ei, const float* __restrict__ a_src,
    const float* __restrict__ a_dst, float* __restrict__ amax) {
  int idx = blockIdx.x * blockDim.x + threadIdx.x;
  if (idx >= E2) return;
  int si = (idx < EE) ? (int)ei[idx]      : (idx - EE);
  int di = (idx < EE) ? (int)ei[EE + idx] : (idx - EE);
  #pragma unroll
  for (int h = 0; h < HEADS; ++h) {
    float v = a_src[si * HEADS + h] + a_dst[di * HEADS + h];
    v = (v > 0.f) ? v : NEG_SLOPE * v;
    atomicMaxFloat(&amax[di * HEADS + h], v);
  }
}

// --------------------------------------------- edge softmax pass 2: denom
__global__ __launch_bounds__(256) void edge_pass_denom(
    const long long* __restrict__ ei, const float* __restrict__ a_src,
    const float* __restrict__ a_dst, const float* __restrict__ amax,
    float* __restrict__ denom) {
  int idx = blockIdx.x * blockDim.x + threadIdx.x;
  if (idx >= E2) return;
  int si = (idx < EE) ? (int)ei[idx]      : (idx - EE);
  int di = (idx < EE) ? (int)ei[EE + idx] : (idx - EE);
  #pragma unroll
  for (int h = 0; h < HEADS; ++h) {
    float v = a_src[si * HEADS + h] + a_dst[di * HEADS + h];
    v = (v > 0.f) ? v : NEG_SLOPE * v;
    atomicAdd(&denom[di * HEADS + h], __expf(v - amax[di * HEADS + h]));
  }
}

// --------------------------- edge pass 3: alpha-weighted, head-averaged sum
// One block (256 threads = feature dim) per edge; folds the /HEADS average in.
__global__ __launch_bounds__(256) void edge_agg(
    const long long* __restrict__ ei, const float* __restrict__ a_src,
    const float* __restrict__ a_dst, const float* __restrict__ amax,
    const float* __restrict__ denom, const float* __restrict__ g,
    float* __restrict__ omean) {
  int idx = blockIdx.x;
  int c   = threadIdx.x;
  int si = (idx < EE) ? (int)ei[idx]      : (idx - EE);
  int di = (idx < EE) ? (int)ei[EE + idx] : (idx - EE);
  __shared__ float alpha[HEADS];
  if (c < HEADS) {
    float v = a_src[si * HEADS + c] + a_dst[di * HEADS + c];
    v = (v > 0.f) ? v : NEG_SLOPE * v;
    alpha[c] = __expf(v - amax[di * HEADS + c]) / denom[di * HEADS + c]
               * (1.0f / HEADS);
  }
  __syncthreads();
  const float* gp = g + (size_t)si * (HEADS * HID);
  float s = alpha[0] * gp[c]
          + alpha[1] * gp[HID + c]
          + alpha[2] * gp[2 * HID + c]
          + alpha[3] * gp[3 * HID + c];
  atomicAdd(&omean[(size_t)di * HID + c], s);
}

// ----------------------------- bias+relu then global mean pool accumulation
__global__ __launch_bounds__(256) void gat_finalize(
    const float* __restrict__ omean, const float* __restrict__ gat_b,
    const long long* __restrict__ batch, float* __restrict__ pooled,
    float* __restrict__ cnt) {
  int n = blockIdx.x, c = threadIdx.x;
  float v = fmaxf(omean[(size_t)n * HID + c] + gat_b[c], 0.0f);
  int b = (int)batch[n];
  atomicAdd(&pooled[b * HID + c], v);
  if (c == 0) atomicAdd(&cnt[b], 1.0f);
}

// ------------------------------------------------------- action head (tiny)
__global__ __launch_bounds__(32) void head_gemm(
    const float* __restrict__ pooled, const float* __restrict__ cnt,
    const float* __restrict__ hw, const float* __restrict__ hb,
    float* __restrict__ out) {
  int b = blockIdx.x, a = threadIdx.x;
  float inv = 1.0f / fmaxf(cnt[b], 1.0f);
  float s = hb[a];
  for (int k = 0; k < HID; ++k)
    s = fmaf(pooled[b * HID + k] * inv, hw[k * ACTIONS + a], s);
  out[b * ACTIONS + a] = s;
}

extern "C" void kernel_launch(void* const* d_in, const int* in_sizes, int n_in,
                              void* d_out, int out_size, void* d_ws, size_t ws_size,
                              hipStream_t stream) {
  (void)in_sizes; (void)n_in; (void)out_size; (void)ws_size;
  const float*     x     = (const float*)d_in[0];
  const long long* ei    = (const long long*)d_in[1];
  const long long* batch = (const long long*)d_in[2];
  const float*     w_l   = (const float*)d_in[3];
  const float*     b_l   = (const float*)d_in[4];
  const float*     w_r   = (const float*)d_in[5];
  const float*     gw    = (const float*)d_in[6];
  const float*     att_s = (const float*)d_in[7];
  const float*     att_d = (const float*)d_in[8];
  const float*     gat_b = (const float*)d_in[9];
  const float*     hw    = (const float*)d_in[10];
  const float*     hb    = (const float*)d_in[11];
  float*           out   = (float*)d_out;

  char* ws = (char*)d_ws;
  size_t off = 0;
  auto take = [&](size_t bytes) -> char* {
    char* p = ws + off;
    off += (bytes + 255) & ~(size_t)255;
    return p;
  };
  float*  agg    = (float*) take((size_t)NN * IN_DIM * 4);
  float*  deg    = (float*) take((size_t)NN * 4);
  __bf16* wlb    = (__bf16*)take((size_t)IN_DIM * HID * 2);
  __bf16* wrb    = (__bf16*)take((size_t)IN_DIM * HID * 2);
  __bf16* gwb    = (__bf16*)take((size_t)HID * HEADS * HID * 2);
  __bf16* hbf    = (__bf16*)take((size_t)NN * HID * 2);
  float*  g      = (float*) take((size_t)NN * HEADS * HID * 4);
  float*  a_src  = (float*) take((size_t)NN * HEADS * 4);
  float*  a_dst  = (float*) take((size_t)NN * HEADS * 4);
  float*  amax   = (float*) take((size_t)NN * HEADS * 4);
  float*  denom  = (float*) take((size_t)NN * HEADS * 4);
  float*  omean  = (float*) take((size_t)NN * HID * 4);
  float*  pooled = (float*) take((size_t)BB * HID * 4);
  float*  cnt    = (float*) take((size_t)BB * 4);

  // ---- init accumulators (workspace is poisoned; must zero each call) ----
  fill_f32<<<(NN * IN_DIM + 255) / 256, 256, 0, stream>>>(agg, NN * IN_DIM, 0.f);
  fill_f32<<<(NN + 255) / 256, 256, 0, stream>>>(deg, NN, 0.f);
  fill_f32<<<(NN * HID + 255) / 256, 256, 0, stream>>>(omean, NN * HID, 0.f);
  fill_f32<<<(NN * HEADS + 255) / 256, 256, 0, stream>>>(amax, NN * HEADS, -__builtin_huge_valf());
  fill_f32<<<(NN * HEADS + 255) / 256, 256, 0, stream>>>(denom, NN * HEADS, 0.f);
  fill_f32<<<(BB * HID + 255) / 256, 256, 0, stream>>>(pooled, BB * HID, 0.f);
  fill_f32<<<1, 64, 0, stream>>>(cnt, BB, 0.f);

  // ---- weights -> bf16 for WMMA B operands ----
  cvt_bf16<<<(IN_DIM * HID + 255) / 256, 256, 0, stream>>>(w_l, wlb, IN_DIM * HID);
  cvt_bf16<<<(IN_DIM * HID + 255) / 256, 256, 0, stream>>>(w_r, wrb, IN_DIM * HID);
  cvt_bf16<<<(HID * HEADS * HID + 255) / 256, 256, 0, stream>>>(gw, gwb, HID * HEADS * HID);

  // ---- SAGE: scatter mean then fused WMMA GEMM + bias + relu ----
  sage_scatter<<<EE, 128, 0, stream>>>(x, ei, agg, deg);
  dim3 gs(NN / 128, HID / 32);
  sage_gemm<<<gs, 256, 0, stream>>>(x, agg, deg, b_l, wlb, wrb, hbf);

  // ---- GAT: projection GEMM, attention dots, edge softmax, aggregation ----
  dim3 gg(NN / 128, (HEADS * HID) / 32);
  gat_gemm<<<gg, 256, 0, stream>>>(hbf, gwb, g);
  gat_att<<<(NN * HEADS * 32 + 255) / 256, 256, 0, stream>>>(g, att_s, att_d, a_src, a_dst);
  edge_pass_max<<<(E2 + 255) / 256, 256, 0, stream>>>(ei, a_src, a_dst, amax);
  edge_pass_denom<<<(E2 + 255) / 256, 256, 0, stream>>>(ei, a_src, a_dst, amax, denom);
  edge_agg<<<E2, HID, 0, stream>>>(ei, a_src, a_dst, amax, denom, g, omean);

  // ---- finalize + pool + action head ----
  gat_finalize<<<NN, HID, 0, stream>>>(omean, gat_b, batch, pooled, cnt);
  head_gemm<<<BB, ACTIONS, 0, stream>>>(pooled, cnt, hw, hb, out);
}
